// RNN_29996051595637
// MI455X (gfx1250) — matrix operand order repaired and checked by reading, not compiled, not run on hardware
//
#include <hip/hip_runtime.h>
#include <hip/hip_bf16.h>

typedef __attribute__((ext_vector_type(16))) _Float16 v16h;
typedef __attribute__((ext_vector_type(8)))  float    v8f;

#define HDIM  1024
#define BATCH 16
#define TSTEPS 500
#define DROP 0.8f

// Hardware tanh if the toolchain exposes it (V_TANH_F32 is a CDNA5 TRANS op);
// fall back to libm otherwise.
__device__ __forceinline__ float fast_tanh(float x) {
#if __has_builtin(__builtin_amdgcn_tanhf)
  return __builtin_amdgcn_tanhf(x);
#elif __has_builtin(__builtin_amdgcn_tanh_f32)
  return __builtin_amdgcn_tanh_f32(x);
#else
  return tanhf(x);
#endif
}

// ---------------- f32 -> f16 conversion with K zero-padding ----------------
__global__ void cvt_pad_f16(const float* __restrict__ src, _Float16* __restrict__ dst,
                            int rows, int K, int Kp) {
  int total = rows * Kp;
  for (int idx = blockIdx.x * blockDim.x + threadIdx.x; idx < total;
       idx += gridDim.x * blockDim.x) {
    int r = idx / Kp;
    int c = idx - r * Kp;
    dst[idx] = (c < K) ? (_Float16)src[(size_t)r * K + c] : (_Float16)0.0f;
  }
}

// ---------------- out[M][1024] = A[M][Kp] * W[1024][Kp]^T + bias ----------------
// One wave per 16x16 output tile; K-loop of v_wmma_f32_16x16x32_f16.
__global__ void proj_gemm(const _Float16* __restrict__ A,   // [M][Kp] f16
                          const _Float16* __restrict__ W,   // [N][Kp] f16 (row-major => B^T)
                          const float*    __restrict__ bias,// [N]
                          float*          __restrict__ out, // [M][N] f32
                          int Mtiles, int Kp) {
  const int N = HDIM;
  const int ntiles = N / 16;
  int wave = (blockIdx.x * blockDim.x + threadIdx.x) >> 5;
  if (wave >= Mtiles * ntiles) return;           // wave-uniform guard (EXEC stays all-1)
  int lane = threadIdx.x & 31;
  int mtile = wave / ntiles;
  int ntile = wave - mtile * ntiles;
  int l16 = lane & 15, hi = lane >> 4;
  int m = mtile * 16 + l16;
  int n = ntile * 16 + l16;

  v8f c;
  float bv = bias[n];
#pragma unroll
  for (int r = 0; r < 8; ++r) c[r] = bv;

  const _Float16* arow = A + (size_t)m * Kp;
  const _Float16* brow = W + (size_t)n * Kp;
#pragma unroll 2
  for (int kb = 0; kb < Kp; kb += 32) {
    v16h a, b;
    const _Float16* ap = arow + kb + 16 * hi;    // A: k = kb + 16*hi + e, row m
    ((uint4*)&a)[0] = *(const uint4*)(ap);
    ((uint4*)&a)[1] = *(const uint4*)(ap + 8);
    b = *(const v16h*)(brow + kb + 16 * hi);     // B: k = kb + 16*hi + e, col n
    c = __builtin_amdgcn_wmma_f32_16x16x32_f16(false, a, false, b, (short)0, c,
                                               false, false);
  }
  float* orow = out + (size_t)(mtile * 16) * N + n;
#pragma unroll
  for (int r = 0; r < 8; ++r) orow[(size_t)(r + 8 * hi) * N] = c[r];
}

// ---------------- one recurrence step: h = tanh(wh_t + h_prev @ U^T) * DROP ----------------
// 16 blocks x 128 threads = 64 waves; wave w owns output columns [16w, 16w+16).
// h_prev (16x1024 f16, 32KB) is staged in LDS once per block.
__global__ void __launch_bounds__(128, 1)
rnn_step(const float*    __restrict__ wh_t,   // [16][1024] f32 (precomputed x-proj + bias)
         const _Float16* __restrict__ U,      // [1024][1024] f16
         const _Float16* __restrict__ h_prev, // [16][1024] f16 (ignored when first)
         _Float16*       __restrict__ h_out,  // [16][1024] f16
         float*          __restrict__ f_out,  // optional [16][1024] f32 (final layer)
         int first) {
  __shared__ _Float16 sh[BATCH * HDIM];        // 32 KB
  int tid = threadIdx.x;
  uint4* s4 = (uint4*)sh;                      // 2048 uint4 total, 16 per thread
  if (first) {
    uint4 z = {0u, 0u, 0u, 0u};
#pragma unroll
    for (int i = 0; i < 16; ++i) s4[tid + i * 128] = z;
  } else {
    const uint4* g4 = (const uint4*)h_prev;
#pragma unroll
    for (int i = 0; i < 16; ++i) s4[tid + i * 128] = g4[tid + i * 128];
  }
  __syncthreads();

  int wave = blockIdx.x * 4 + (tid >> 5);      // 0..63 = ntile
  int lane = tid & 31;
  int l16 = lane & 15, hi = lane >> 4;
  int n = wave * 16 + l16;

  v8f c;
#pragma unroll
  for (int r = 0; r < 8; ++r) c[r] = wh_t[(size_t)(r + 8 * hi) * HDIM + n];

  const _Float16* srow = sh + l16 * HDIM;      // A row m = lane%16 from LDS
  const _Float16* brow = U + (size_t)n * HDIM; // B^T row n from global (L2-hot)
#pragma unroll 4
  for (int kb = 0; kb < HDIM; kb += 32) {
    v16h a, b;
    const _Float16* ap = srow + kb + 16 * hi;
    ((uint4*)&a)[0] = *(const uint4*)(ap);
    ((uint4*)&a)[1] = *(const uint4*)(ap + 8);
    b = *(const v16h*)(brow + kb + 16 * hi);
    c = __builtin_amdgcn_wmma_f32_16x16x32_f16(false, a, false, b, (short)0, c,
                                               false, false);
  }

  _Float16* ho = h_out + n;
#pragma unroll
  for (int r = 0; r < 8; ++r) {
    float h = DROP * fast_tanh(c[r]);
    ho[(size_t)(r + 8 * hi) * HDIM] = (_Float16)h;
    if (f_out) f_out[(size_t)(r + 8 * hi) * HDIM + n] = h;
  }
}

// ---------------- host orchestration ----------------
extern "C" void kernel_launch(void* const* d_in, const int* in_sizes, int n_in,
                              void* d_out, int out_size, void* d_ws, size_t ws_size,
                              hipStream_t stream) {
  const float* x  = (const float*)d_in[0];   // [500][16][440]
  const float* w0 = (const float*)d_in[1];   // [1024][440]
  const float* b0 = (const float*)d_in[2];   // [1024]
  const float* u0 = (const float*)d_in[3];   // [1024][1024]
  const float* w1 = (const float*)d_in[4];   // [1024][1024]
  const float* b1 = (const float*)d_in[5];   // [1024]
  const float* u1 = (const float*)d_in[6];   // [1024][1024]
  float* out = (float*)d_out;                // [500][16][1024]

  const int T = TSTEPS, B = BATCH, F = 440, H = HDIM, Fp = 448;
  const int M = T * B;  // 8000 rows

  char* ws = (char*)d_ws;
  size_t off = 0;
  auto alloc = [&](size_t bytes) -> char* {
    char* p = ws + off;
    off = (off + bytes + 255) & ~(size_t)255;
    return p;
  };
  float*    proj = (float*)   alloc((size_t)M * H * 4);  // 31.25 MB, reused both layers
  _Float16* hf   = (_Float16*)alloc((size_t)M * H * 2);  // 15.6 MB hidden states f16
  _Float16* xh   = (_Float16*)alloc((size_t)M * Fp * 2);
  _Float16* w0h  = (_Float16*)alloc((size_t)H * Fp * 2);
  _Float16* u0h  = (_Float16*)alloc((size_t)H * H * 2);
  _Float16* w1h  = (_Float16*)alloc((size_t)H * H * 2);
  _Float16* u1h  = (_Float16*)alloc((size_t)H * H * 2);
  (void)ws_size; (void)in_sizes; (void)n_in; (void)out_size;

  // One-time f32->f16 conversions (weights L2-resident thereafter)
  cvt_pad_f16<<<2048, 256, 0, stream>>>(x,  xh,  M, F, Fp);
  cvt_pad_f16<<<1024, 256, 0, stream>>>(w0, w0h, H, F, Fp);
  cvt_pad_f16<<<2048, 256, 0, stream>>>(u0, u0h, H, H, H);
  cvt_pad_f16<<<2048, 256, 0, stream>>>(w1, w1h, H, H, H);
  cvt_pad_f16<<<2048, 256, 0, stream>>>(u1, u1h, H, H, H);

  // Layer 0: input projection (8000x448x1024), then 500 serial steps
  {
    int waves  = (M / 16) * (H / 16);          // 32000 tiles
    int blocks = (waves * 32 + 255) / 256;     // 4000 blocks of 8 waves
    proj_gemm<<<blocks, 256, 0, stream>>>(xh, w0h, b0, proj, M / 16, Fp);
  }
  for (int t = 0; t < T; ++t) {
    const _Float16* hp = (t == 0) ? hf : hf + (size_t)(t - 1) * B * H;
    rnn_step<<<16, 128, 0, stream>>>(proj + (size_t)t * B * H, u0h, hp,
                                     hf + (size_t)t * B * H, nullptr, t == 0);
  }

  // Layer 1: projection of layer-0 hidden states (8000x1024x1024), then 500 steps.
  // proj buffer is reused; hf is overwritten in-place by layer-1 recurrence (safe:
  // step t only needs proj[t] and layer-1's own h[t-1]).
  {
    int waves  = (M / 16) * (H / 16);
    int blocks = (waves * 32 + 255) / 256;
    proj_gemm<<<blocks, 256, 0, stream>>>(hf, w1h, b1, proj, M / 16, H);
  }
  for (int t = 0; t < T; ++t) {
    const _Float16* hp = (t == 0) ? hf : hf + (size_t)(t - 1) * B * H;
    rnn_step<<<16, 128, 0, stream>>>(proj + (size_t)t * B * H, u1h, hp,
                                     hf + (size_t)t * B * H,
                                     out + (size_t)t * B * H, t == 0);
  }
}
